// WaveNetModel_39917426049109
// MI455X (gfx1250) — matrix operand order repaired and checked by reading, not compiled, not run on hardware
//
#include <hip/hip_runtime.h>
#include <hip/hip_bf16.h>
#include <math.h>

// ---------------------------------------------------------------------------
// WaveNet forward for MI455X (gfx1250), bf16 WMMA path.
//   h buffers  : bf16 in workspace (ping-pong)
//   skip accum : fp32 in workspace
//   weights    : packed bf16 A-fragments (16x32 tiles, 32 lanes x 16 bf16)
//   layer tile : 64 time steps x 256 channels, dynamic LDS (84 KB, WGP has 320 KB)
// ---------------------------------------------------------------------------

typedef __attribute__((ext_vector_type(16))) __bf16 v16bf;
typedef __attribute__((ext_vector_type(8)))  float  v8f;

#define NGATED 39
#define LF     4612          // final length: 8192 - (3*1023 + 511)
#define TN     64            // time-tile per block
#define LDSK   144           // padded K stride (288 B = 9*32 -> 32B-aligned frags)
#define ZBYTES (256 * TN * 4)                       // 65536
#define LDSBYTES (ZBYTES + TN * LDSK * 2)           // 65536 + 18432 = 83968

__device__ __forceinline__ unsigned short f2bf(float f) {
  union { float f; unsigned int u; } v; v.f = f;
  unsigned int r = v.u + 0x7FFFu + ((v.u >> 16) & 1u);   // RNE
  return (unsigned short)(r >> 16);
}
__device__ __forceinline__ float bf2f(unsigned short b) {
  union { unsigned int u; float f; } v; v.u = ((unsigned int)b) << 16;
  return v.f;
}

// (m,k) of slot j for lane, inside one 16x32 bf16 A tile (contiguous k = 16*hi + j)
__device__ __forceinline__ void a_frag_mk(int kt, int lane, int j, int& m, int& k) {
  const int hi = lane >> 4;
  m = lane & 15;
  const int kk = hi * 16 + j;
  k = kt * 32 + kk;
}

// ---------------------------------------------------------------------------
// Pack fp32 weights into bf16 WMMA A-fragment order.
//   dil : [39][2 taps][16 mt][4 kt][32 lane][16]   (65536 elems / layer)
//   skip: [39][16 mt][4 kt][32][16]                (32768 / layer)
//   res : [39][ 8 mt][4 kt][32][16]                (16384 / layer)
// ---------------------------------------------------------------------------
__global__ __launch_bounds__(256) void pack_weights(
    const float* __restrict__ dilW, const float* __restrict__ skipW,
    const float* __restrict__ resW,
    unsigned short* __restrict__ dilP, unsigned short* __restrict__ skipP,
    unsigned short* __restrict__ resP)
{
  const int DIL_T = NGATED * 65536;
  const int SKP_T = NGATED * 32768;
  const int RES_T = NGATED * 16384;
  int gid = blockIdx.x * 256 + threadIdx.x;
  if (gid < DIL_T) {
    int layer = gid >> 16, r = gid & 65535;
    int tap = r >> 15; r &= 32767;
    int mt = r >> 11, kt = (r >> 9) & 3, lane = (r >> 4) & 31, j = r & 15;
    int m, k; a_frag_mk(kt, lane, j, m, k); m += mt * 16;
    dilP[gid] = f2bf(dilW[(((size_t)layer * 256 + m) * 128 + k) * 2 + tap]);
    return;
  }
  gid -= DIL_T;
  if (gid < SKP_T) {
    int layer = gid >> 15, r = gid & 32767;
    int mt = r >> 11, kt = (r >> 9) & 3, lane = (r >> 4) & 31, j = r & 15;
    int m, k; a_frag_mk(kt, lane, j, m, k); m += mt * 16;
    skipP[gid] = f2bf(skipW[((size_t)layer * 256 + m) * 128 + k]);
    return;
  }
  gid -= SKP_T;
  if (gid < RES_T) {
    int layer = gid >> 14, r = gid & 16383;
    int mt = r >> 11, kt = (r >> 9) & 3, lane = (r >> 4) & 31, j = r & 15;
    int m, k; a_frag_mk(kt, lane, j, m, k); m += mt * 16;
    resP[gid] = f2bf(resW[((size_t)layer * 128 + m) * 128 + k]);
  }
}

// ---------------------------------------------------------------------------
// Start 1x1 conv: 40 -> 128, fp32 in, bf16 out. Memory bound; plain VALU.
// ---------------------------------------------------------------------------
__global__ __launch_bounds__(256) void start_conv(
    const float* __restrict__ x, const float* __restrict__ w,
    const float* __restrict__ b, unsigned short* __restrict__ h)
{
  const size_t gid = (size_t)blockIdx.x * 256 + threadIdx.x;
  if (gid >= (size_t)8 * 128 * 8192) return;
  const int t = (int)(gid & 8191);
  const int c = (int)((gid >> 13) & 127);
  const int n = (int)(gid >> 20);
  float s = b[c];
  const float* xp = x + (size_t)n * 40 * 8192 + t;
  const float* wp = w + c * 40;
#pragma unroll 8
  for (int ci = 0; ci < 40; ++ci) s += wp[ci] * xp[(size_t)ci * 8192];
  h[gid] = f2bf(s);
}

// ---------------------------------------------------------------------------
// End 1x1 conv: relu(skip[256]) -> 64, fp32 out.
// ---------------------------------------------------------------------------
__global__ __launch_bounds__(256) void end_conv(
    const float* __restrict__ skip, const float* __restrict__ w,
    const float* __restrict__ b, float* __restrict__ out)
{
  const size_t total = (size_t)8 * 64 * LF;
  const size_t gid = (size_t)blockIdx.x * 256 + threadIdx.x;
  if (gid >= total) return;
  const int t = (int)(gid % LF);
  const int c = (int)((gid / LF) % 64);
  const int n = (int)(gid / ((size_t)64 * LF));
  float s = b[c];
  const float* sp = skip + (size_t)n * 256 * LF + t;
  const float* wp = w + c * 256;
  for (int ci = 0; ci < 256; ++ci) {
    const float v = sp[(size_t)ci * LF];
    s += wp[ci] * (v > 0.f ? v : 0.f);
  }
  out[gid] = s;
}

// ---------------------------------------------------------------------------
// Fused gated layer. One block = one batch n, 64 time steps, dynamic LDS.
//   GEMM1 : z[256 x 64] = Wd0 @ h[t] + Wd1 @ h[t+d]     (bf16 WMMA, f32 acc)
//   gate  : a = tanh(f) * sigmoid(g)                    (128 x 64)
//   GEMM2a: skipAcc += Wskip @ a + bskip  (tail-aligned fp32 RMW)
//   GEMM2b: h' = Wres @ a + bres + h[t+d]               (bf16 out)
// LDS map: [0, 36864)  stage[2][TN][LDSK] bf16   (live for GEMM1)
//          [0, 65536)  z[256][TN] f32            (aliases stage after GEMM1)
//          [65536, 83968) atile[TN][LDSK] bf16   (gated activation)
// ---------------------------------------------------------------------------
__global__ __launch_bounds__(256) void wavenet_layer(
    const unsigned short* __restrict__ hin, unsigned short* __restrict__ hout,
    const unsigned short* __restrict__ wDil, const unsigned short* __restrict__ wSkip,
    const unsigned short* __restrict__ wRes,
    const float* __restrict__ bDil, const float* __restrict__ bSkip,
    const float* __restrict__ bRes,
    float* __restrict__ skipAcc, int Lin, int Lout, int dil)
{
  extern __shared__ char smem[];
  unsigned short* stage = (unsigned short*)smem;            // [2][TN][LDSK]
  float*          zbuf  = (float*)smem;                     // [256][TN]
  unsigned short* atile = (unsigned short*)(smem + ZBYTES); // [TN][LDSK]

  const int tid  = threadIdx.x;
  const int lane = tid & 31;
  const int wave = tid >> 5;               // 8 wave32 per block
  const int n    = blockIdx.y;
  const int t0   = blockIdx.x * TN;
  const size_t hinN = (size_t)n * 128 * (size_t)Lin;

  const v8f vzero = {0.f,0.f,0.f,0.f,0.f,0.f,0.f,0.f};

  // Prefetch phase-2 weights while GEMM1 runs (global_prefetch_b8).
  __builtin_prefetch(wSkip + (size_t)tid * 128, 0, 1);
  __builtin_prefetch(wRes  + (size_t)tid * 64,  0, 1);

  // ---- stage both taps of h into LDS, transposed to [t][c] ----
  for (int tap = 0; tap < 2; ++tap) {
    const int base = t0 + tap * dil;
    for (int idx = tid; idx < 128 * (TN / 8); idx += 256) {
      const int c  = idx / (TN / 8);
      const int j8 = (idx % (TN / 8)) * 8;
      const unsigned short* src = hin + hinN + (size_t)c * Lin + base + j8;
#pragma unroll
      for (int e = 0; e < 8; ++e) {
        const int pos = base + j8 + e;
        stage[((tap * TN + j8 + e) * LDSK) + c] =
            (pos < Lin) ? src[e] : (unsigned short)0;
      }
    }
  }
  __syncthreads();

  const int mt0  = wave * 2;            // two 16-row M-tiles per wave (z: 256 rows)
  const int rowb = lane & 15;           // B fragment: N = lane%16
  const int colb = (lane >> 4) << 4;    // B fragment: +16 K for hi half
  const int mrow = (lane >> 4) << 3;    // C/D fragment: +8 M for hi half

  // ---- GEMM1: dilated conv (2 taps x 4 K-steps x 4 N-tiles) ----
  v8f acc[2][4] = {{vzero, vzero, vzero, vzero}, {vzero, vzero, vzero, vzero}};
#pragma unroll
  for (int tap = 0; tap < 2; ++tap) {
#pragma unroll
    for (int kt = 0; kt < 4; ++kt) {
      const v16bf a0 = *(const v16bf*)(wDil + (((size_t)(tap*16 + mt0    ) * 4 + kt) << 9) + lane*16);
      const v16bf a1 = *(const v16bf*)(wDil + (((size_t)(tap*16 + mt0 + 1) * 4 + kt) << 9) + lane*16);
#pragma unroll
      for (int nt = 0; nt < 4; ++nt) {
        const v16bf bb = *(const v16bf*)(&stage[(tap * TN + nt*16 + rowb) * LDSK + kt*32 + colb]);
        acc[0][nt] = __builtin_amdgcn_wmma_f32_16x16x32_bf16(false, a0, false, bb, (short)0, acc[0][nt], false, false);
        acc[1][nt] = __builtin_amdgcn_wmma_f32_16x16x32_bf16(false, a1, false, bb, (short)0, acc[1][nt], false, false);
      }
    }
  }
  __syncthreads();  // stage dead; z may alias it now

  // scatter f32 accumulators into z (C/D VGPR layout: VGPR r -> M=r or r+8)
#pragma unroll
  for (int mi = 0; mi < 2; ++mi)
#pragma unroll
    for (int nt = 0; nt < 4; ++nt)
#pragma unroll
      for (int r = 0; r < 8; ++r)
        zbuf[((mt0 + mi) * 16 + mrow + r) * TN + nt * 16 + rowb] = acc[mi][nt][r];
  __syncthreads();

  // ---- gated activation, restage as bf16 [t][c] ----
  for (int idx = tid; idx < 128 * TN; idx += 256) {
    const int c = idx / TN, j = idx % TN;
    const float f = zbuf[c * TN + j]         + bDil[c];
    const float g = zbuf[(c + 128) * TN + j] + bDil[c + 128];
    const float av = tanhf(f) * (1.0f / (1.0f + __expf(-g)));
    atile[j * LDSK + c] = f2bf(av);
  }
  __syncthreads();

  // ---- GEMM2a: skip conv (256 rows), accumulate fp32 tail-aligned ----
  v8f sacc[2][4] = {{vzero, vzero, vzero, vzero}, {vzero, vzero, vzero, vzero}};
#pragma unroll
  for (int kt = 0; kt < 4; ++kt) {
    const v16bf a0 = *(const v16bf*)(wSkip + (((size_t)(mt0    ) * 4 + kt) << 9) + lane*16);
    const v16bf a1 = *(const v16bf*)(wSkip + (((size_t)(mt0 + 1) * 4 + kt) << 9) + lane*16);
#pragma unroll
    for (int nt = 0; nt < 4; ++nt) {
      const v16bf bb = *(const v16bf*)(&atile[(nt*16 + rowb) * LDSK + kt*32 + colb]);
      sacc[0][nt] = __builtin_amdgcn_wmma_f32_16x16x32_bf16(false, a0, false, bb, (short)0, sacc[0][nt], false, false);
      sacc[1][nt] = __builtin_amdgcn_wmma_f32_16x16x32_bf16(false, a1, false, bb, (short)0, sacc[1][nt], false, false);
    }
  }
  const int soff = Lout - LF;   // >= 0 for every layer
#pragma unroll
  for (int mi = 0; mi < 2; ++mi)
#pragma unroll
    for (int nt = 0; nt < 4; ++nt)
#pragma unroll
      for (int r = 0; r < 8; ++r) {
        const int c = (mt0 + mi) * 16 + mrow + r;
        const int t = t0 + nt * 16 + rowb;
        const int sp = t - soff;
        if (t < Lout && sp >= 0) {
          float* p = skipAcc + ((size_t)n * 256 + c) * LF + sp;
          *p += sacc[mi][nt][r] + bSkip[c];   // unique (n,c,sp) per layer: plain RMW
        }
      }

  // ---- GEMM2b: res conv (128 rows) + residual + bf16 store ----
  v8f racc[4] = {vzero, vzero, vzero, vzero};
#pragma unroll
  for (int kt = 0; kt < 4; ++kt) {
    const v16bf a0 = *(const v16bf*)(wRes + (((size_t)wave * 4 + kt) << 9) + lane*16);
#pragma unroll
    for (int nt = 0; nt < 4; ++nt) {
      const v16bf bb = *(const v16bf*)(&atile[(nt*16 + rowb) * LDSK + kt*32 + colb]);
      racc[nt] = __builtin_amdgcn_wmma_f32_16x16x32_bf16(false, a0, false, bb, (short)0, racc[nt], false, false);
    }
  }
#pragma unroll
  for (int nt = 0; nt < 4; ++nt)
#pragma unroll
    for (int r = 0; r < 8; ++r) {
      const int c = wave * 16 + mrow + r;
      const int t = t0 + nt * 16 + rowb;
      if (t < Lout) {
        const float resid = bf2f(hin[hinN + (size_t)c * Lin + t + dil]);
        const float v = racc[nt][r] + bRes[c] + resid;
        hout[((size_t)n * 128 + c) * (size_t)Lout + t] = f2bf(v);
      }
    }
}

// ---------------------------------------------------------------------------
// Host launch: memset skip, pack weights, start conv, 39 fused layers, end.
// ---------------------------------------------------------------------------
extern "C" void kernel_launch(void* const* d_in, const int* in_sizes, int n_in,
                              void* d_out, int out_size, void* d_ws, size_t ws_size,
                              hipStream_t stream)
{
  const float* x      = (const float*)d_in[0];
  const float* startW = (const float*)d_in[1];
  const float* startB = (const float*)d_in[2];
  const float* dilW   = (const float*)d_in[3];
  const float* dilB   = (const float*)d_in[4];
  const float* resW   = (const float*)d_in[5];
  const float* resB   = (const float*)d_in[6];
  const float* skipW  = (const float*)d_in[7];
  const float* skipB  = (const float*)d_in[8];
  const float* endW   = (const float*)d_in[9];
  const float* endB   = (const float*)d_in[10];
  float* out = (float*)d_out;

  char* ws = (char*)d_ws;
  size_t off = 0;
  auto take = [&](size_t bytes) -> char* {
    char* p = ws + off;
    off = (off + bytes + 255) & ~(size_t)255;
    return p;
  };
  unsigned short* hA      = (unsigned short*)take((size_t)8 * 128 * 8192 * 2);
  unsigned short* hB      = (unsigned short*)take((size_t)8 * 128 * 8192 * 2);
  float*          skipAcc = (float*)take((size_t)8 * 256 * LF * 4);
  unsigned short* dilP    = (unsigned short*)take((size_t)NGATED * 65536 * 2);
  unsigned short* skipP   = (unsigned short*)take((size_t)NGATED * 32768 * 2);
  unsigned short* resP    = (unsigned short*)take((size_t)NGATED * 16384 * 2);

  hipMemsetAsync(skipAcc, 0, (size_t)8 * 256 * LF * 4, stream);

  {
    const int total = NGATED * (65536 + 32768 + 16384);
    pack_weights<<<(total + 255) / 256, 256, 0, stream>>>(dilW, skipW, resW, dilP, skipP, resP);
  }
  {
    const size_t total = (size_t)8 * 128 * 8192;
    start_conv<<<(unsigned)((total + 255) / 256), 256, 0, stream>>>(x, startW, startB, hA);
  }

  static const int DIL[NGATED] = {
    1, 2, 4, 8, 16, 32, 64, 128, 256, 512,
    1, 2, 4, 8, 16, 32, 64, 128, 256, 512,
    1, 2, 4, 8, 16, 32, 64, 128, 256, 512,
    1, 2, 4, 8, 16, 32, 64, 128, 256
  };

  int Lin = 8192;
  unsigned short* hin = hA;
  unsigned short* hout = hB;
  for (int i = 0; i < NGATED; ++i) {
    const int d = DIL[i];
    const int Lout = Lin - d;
    dim3 grid((Lout + TN - 1) / TN, 8);
    wavenet_layer<<<grid, 256, LDSBYTES, stream>>>(
        hin, hout,
        dilP + (size_t)i * 65536, skipP + (size_t)i * 32768, resP + (size_t)i * 16384,
        dilB + (size_t)i * 256, skipB + (size_t)i * 256, resB + (size_t)i * 128,
        skipAcc, Lin, Lout, d);
    unsigned short* tmp = hin; hin = hout; hout = tmp;
    Lin = Lout;
  }
  {
    const size_t total = (size_t)8 * 64 * LF;
    end_conv<<<(unsigned)((total + 255) / 256), 256, 0, stream>>>(skipAcc, endW, endB, out);
  }
}
